// CRFLayer_4458176053273
// MI455X (gfx1250) — matrix operand order repaired and checked by reading, not compiled, not run on hardware
//
#include <hip/hip_runtime.h>

typedef _Float16 v8h  __attribute__((ext_vector_type(8)));
typedef _Float16 v16h __attribute__((ext_vector_type(16)));
typedef float    v8f  __attribute__((ext_vector_type(8)));
typedef unsigned int v4u __attribute__((ext_vector_type(4)));
typedef int          v4i __attribute__((ext_vector_type(4)));
typedef int          v8i __attribute__((ext_vector_type(8)));

#define RTAGS     128        // REAL = TAG_SIZE + 2
#define START_TAG 126
#define END_TAG   127
#define NBATCH    256
#define SEQLEN    1024
#define NSLOT     4          // LDS ring slots for TDM-prefetched emit tiles
#define TDMD      3          // TDM prefetch distance (NSLOT-1)

#define LOG2E 1.4426950408889634f
#define LN2   0.6931471805599453f

__device__ __forceinline__ float fast_ln(float x)  { return __builtin_amdgcn_logf(x) * LN2; }
__device__ __forceinline__ float fast_exp(float x) { return __builtin_amdgcn_exp2f(x * LOG2E); }

// Cross-lane max over each 16-lane group: 4 single-instruction DPP16 stages.
__device__ __forceinline__ float rowmax16(float v) {
    float t;
    asm("v_max_num_f32_dpp %0, %1, %1 quad_perm:[1,0,3,2] row_mask:0xf bank_mask:0xf bound_ctrl:1"
        : "=v"(t) : "v"(v));
    asm("v_max_num_f32_dpp %0, %1, %1 quad_perm:[2,3,0,1] row_mask:0xf bank_mask:0xf bound_ctrl:1"
        : "=v"(v) : "v"(t));
    asm("v_max_num_f32_dpp %0, %1, %1 row_half_mirror row_mask:0xf bank_mask:0xf bound_ctrl:1"
        : "=v"(t) : "v"(v));
    asm("v_max_num_f32_dpp %0, %1, %1 row_mirror row_mask:0xf bank_mask:0xf bound_ctrl:1"
        : "=v"(v) : "v"(t));
    return v;
}

// One TDM descriptor: 2D tile, 16 rows x 128 f32, row stride 512KB -> LDS.
// D# layout per CDNA5 ISA ch.8: group0 = {flags, lds_addr, gaddr_lo, gaddr_hi|type=2},
// group1 = {data_size=4B, tensor_dim0=128, tensor_dim1=16, tile_dim0=128,
//           tile_dim1=16, dim0_stride=131072, 0, 0}; groups 2/3 zero (2-D).
__device__ __forceinline__ void tdm_load_tile(const float* gaddr, unsigned ldsOff) {
    unsigned long long ga = (unsigned long long)(uintptr_t)gaddr;
    v4u g0 = { 1u,                                  // count=1 (valid), load, user
               ldsOff,                              // lds_addr (bytes)
               (unsigned)ga,                        // global_addr[31:0]
               (unsigned)(ga >> 32) | 0x80000000u };// global_addr[56:32] | type=2
    v8i g1 = { (int)0x00020000,       // data_size = 2 (4 bytes)
               (int)(128u << 16),     // tensor_dim0 = 128 (low 16 in [31:16])
               (int)(16u << 16),      // tensor_dim0 hi=0 | tensor_dim1 = 16
               (int)(128u << 16),     // tensor_dim1 hi=0 | tile_dim0 = 128
               16,                    // tile_dim1 = 16 | tile_dim2 = 0
               131072,                // tensor_dim0_stride (data_size units)
               0, 0 };
    v4i g2 = {0, 0, 0, 0};
    v4i g3 = {0, 0, 0, 0};
    v8i gx = {0, 0, 0, 0, 0, 0, 0, 0};   // 6-arg toolchain: extra group, unused
    __builtin_amdgcn_tensor_load_to_lds(g0, g1, g2, g3, gx, 0);
}

// ---------------------------------------------------------------------------
// Gold path score: gold[b] = sum_t trans[y_t, y_{t-1}] + X[b,t,y_t]
//                            + trans[END, y_{L-1}]
// ---------------------------------------------------------------------------
__launch_bounds__(256)
__global__ void crf_gold_kernel(const float* __restrict__ X,
                                const int*   __restrict__ y,
                                const float* __restrict__ trans,
                                float*       __restrict__ gold)
{
    const int b   = blockIdx.x;
    const int tid = threadIdx.x;
    const int*   yb = y + (size_t)b * SEQLEN;
    const float* Xb = X + (size_t)b * SEQLEN * RTAGS;

    float s = 0.0f;
    for (int t = tid; t < SEQLEN; t += 256) {
        int yt = yb[t];
        int yp = (t == 0) ? START_TAG : yb[t - 1];
        s += trans[yt * RTAGS + yp] + Xb[(size_t)t * RTAGS + yt];
    }
    s += __shfl_xor(s, 1, 32);
    s += __shfl_xor(s, 2, 32);
    s += __shfl_xor(s, 4, 32);
    s += __shfl_xor(s, 8, 32);
    s += __shfl_xor(s, 16, 32);

    __shared__ float red[8];
    if ((tid & 31) == 0) red[tid >> 5] = s;
    __syncthreads();
    if (tid == 0) {
        float tot = 0.0f;
        #pragma unroll
        for (int i = 0; i < 8; ++i) tot += red[i];
        tot += trans[END_TAG * RTAGS + yb[SEQLEN - 1]];
        gold[b] = tot;
    }
}

// ---------------------------------------------------------------------------
// Forward algorithm: one workgroup = 16 batch rows over all 1024 steps.
// Emits streamed by the Tensor Data Mover into a 4-slot LDS ring (8KB tiles),
// 3 steps ahead; GEMM S = P x E^T per step via v_wmma_f32_16x16x32_f16;
// rescale by previous step's row max (one barrier per step).
// ---------------------------------------------------------------------------
__launch_bounds__(256, 1)
__global__ void crf_forward_kernel(const float* __restrict__ X,
                                   const float* __restrict__ trans,
                                   const float* __restrict__ gold,
                                   float*       __restrict__ out)
{
    // Pool: E=exp(trans) (f16, 32KB) during setup; X-ring (4x8KB) in the loop.
    __shared__ __align__(16) unsigned char ldsPool[NSLOT * 16 * RTAGS * 4];
    __shared__ _Float16 ldsP[2][16 * RTAGS];     // ping-pong P[m][j]
    __shared__ float    ldsMax[2][8 * 16];       // ping-pong per-wave row maxima
    __shared__ float    ldsSum[8 * 16];
    __shared__ float    ldsCarry[16];

    const int tid  = threadIdx.x;
    const int lane = tid & 31;
    const int w    = tid >> 5;      // wave 0..7
    const int n    = lane & 15;     // column within this wave's 16-wide tile
    const int h    = lane >> 4;     // lane half
    const int b0   = blockIdx.x * 16;
    const int col  = w * 16 + n;    // global tag column owned by this lane

    _Float16*    ldsE = (_Float16*)ldsPool;      // setup-only view
    const float* ldsX = (const float*)ldsPool;   // steady-state ring view
    const unsigned poolOff = (unsigned)(uintptr_t)ldsPool;  // LDS byte offset

    // Build E = exp(trans) in LDS (f16). Masked entries (-1e4) -> exactly 0.
    for (int idx = tid; idx < RTAGS * RTAGS; idx += 256)
        ldsE[idx] = (_Float16)fast_exp(trans[idx]);
    for (int idx = tid; idx < 16 * RTAGS; idx += 256) {
        int j = idx & (RTAGS - 1);
        ldsP[0][idx] = (j == START_TAG) ? (_Float16)1.0f : (_Float16)0.0f;
    }
    for (int idx = tid; idx < 2 * 8 * 16; idx += 256)
        ldsMax[0][idx] = 0.0f;
    __syncthreads();

    // Preload constant B fragments (E^T tiles) into registers: 4 K-chunks.
    v16h bfrag[4];
    #pragma unroll
    for (int c = 0; c < 4; ++c) {
        const _Float16* p = &ldsE[col * RTAGS + c * 32 + h * 8];
        v8h lo = *(const v8h*)(p);
        v8h hi = *(const v8h*)(p + 16);
        bfrag[c] = __builtin_shufflevector(lo, hi, 0, 1, 2, 3, 4, 5, 6, 7,
                                                   8, 9, 10, 11, 12, 13, 14, 15);
    }
    const float tEndc = trans[END_TAG * RTAGS + col];
    __syncthreads();   // all E reads done: pool may be overwritten by the TDM

    // Prologue: DMA tiles for steps 0..TDMD-1 into ring slots 0..TDMD-1.
    if (w == 0) {
        #pragma unroll
        for (int d = 0; d < TDMD; ++d)
            tdm_load_tile(X + ((size_t)b0 * SEQLEN + d) * RTAGS,
                          poolOff + d * (16 * RTAGS * 4));
        __builtin_amdgcn_s_wait_tensorcnt(TDMD - 1);   // slot 0 resident
    }
    __syncthreads();

    float carry8[8];
    float anorm[8];
    #pragma unroll
    for (int r = 0; r < 8; ++r) { carry8[r] = 0.0f; anorm[r] = 0.0f; }

    for (int tt = 0; tt < SEQLEN; tt += 8) {
        #pragma unroll
        for (int u = 0; u < 8; ++u) {
            const int t    = tt + u;
            const int pc   = u & 1;              // P buffer read / own-max write
            const int pn   = pc ^ 1;             // P buffer write / prev-max read
            const int slot = u & (NSLOT - 1);    // ring slot for this step

            // --- TDM: issue tile t+TDMD into slot (t-1)&3 (consumed last step)
            if (w == 0) {
                int tn = (t + TDMD) & (SEQLEN - 1);   // wrap: in-bounds dummy tail
                tdm_load_tile(X + ((size_t)b0 * SEQLEN + tn) * RTAGS,
                              poolOff + ((u + TDMD) & (NSLOT - 1)) * (16 * RTAGS * 4));
            }

            // --- emits for this step from the resident ring slot ----------
            float xv8[8];
            #pragma unroll
            for (int r = 0; r < 8; ++r)
                xv8[r] = ldsX[slot * (16 * RTAGS) + (8 * h + r) * RTAGS + col];

            // --- A fragments: 8x ds_load_b128 -----------------------------
            v16h afrag[4];
            #pragma unroll
            for (int c = 0; c < 4; ++c) {
                const _Float16* p = &ldsP[pc][n * RTAGS + c * 32 + h * 8];
                v8h lo = *(const v8h*)(p);
                v8h hi = *(const v8h*)(p + 16);
                afrag[c] = __builtin_shufflevector(lo, hi, 0, 1, 2, 3, 4, 5, 6, 7,
                                                           8, 9, 10, 11, 12, 13, 14, 15);
            }

            // --- GEMM, two independent accumulator chains -----------------
            v8f acc0 = {0.f, 0.f, 0.f, 0.f, 0.f, 0.f, 0.f, 0.f};
            v8f acc1 = {0.f, 0.f, 0.f, 0.f, 0.f, 0.f, 0.f, 0.f};
            acc0 = __builtin_amdgcn_wmma_f32_16x16x32_f16(false, afrag[0], false, bfrag[0],
                                                          (short)0, acc0, false, false);
            acc1 = __builtin_amdgcn_wmma_f32_16x16x32_f16(false, afrag[1], false, bfrag[1],
                                                          (short)0, acc1, false, false);
            acc0 = __builtin_amdgcn_wmma_f32_16x16x32_f16(false, afrag[2], false, bfrag[2],
                                                          (short)0, acc0, false, false);
            acc1 = __builtin_amdgcn_wmma_f32_16x16x32_f16(false, afrag[3], false, bfrag[3],
                                                          (short)0, acc1, false, false);

            // --- alpha' = ln(S) + emit ------------------------------------
            float av[8];
            #pragma unroll
            for (int r = 0; r < 8; ++r)
                av[r] = fast_ln(acc0[r] + acc1[r]) + xv8[r];

            // --- in-wave row max (DPP); publish for step t+1 --------------
            float rmax8[8];
            #pragma unroll
            for (int r = 0; r < 8; ++r)
                rmax8[r] = rowmax16(av[r]);
            if (n == 0) {   // single guarded block: 8 contiguous stores
                #pragma unroll
                for (int r = 0; r < 8; ++r)
                    ldsMax[pc][w * 16 + 8 * h + r] = rmax8[r];
            }

            // --- scale with PREVIOUS step's cross-wave max ----------------
            #pragma unroll
            for (int r = 0; r < 8; ++r) {
                float mh = ldsMax[pn][8 * h + r];
                #pragma unroll
                for (int ww = 1; ww < 8; ++ww)
                    mh = fmaxf(mh, ldsMax[pn][ww * 16 + 8 * h + r]);
                float a2 = av[r] - mh;
                anorm[r]  = a2;
                carry8[r] += mh;
                // f16 overflow guard: drift vs prev-step scale is bounded.
                ldsP[pn][(8 * h + r) * RTAGS + col] =
                    (_Float16)fast_exp(fminf(a2, 11.0f));
            }

            // --- ensure tile for step t+1 resident, then step barrier -----
            if (w == 0) __builtin_amdgcn_s_wait_tensorcnt(TDMD - 1);
            __syncthreads();
        }
    }

    // --- logZ = carry + ln sum_i exp(anorm_i + trans[END,i]) ------------
    float sv[8];
    #pragma unroll
    for (int r = 0; r < 8; ++r) {
        float v = fast_exp(anorm[r] + tEndc);
        v += __shfl_xor(v, 1, 32);
        v += __shfl_xor(v, 2, 32);
        v += __shfl_xor(v, 4, 32);
        v += __shfl_xor(v, 8, 32);
        sv[r] = v;
    }
    if (n == 0) {
        #pragma unroll
        for (int r = 0; r < 8; ++r)
            ldsSum[w * 16 + 8 * h + r] = sv[r];
    }
    if (w == 0 && n == 0) {
        #pragma unroll
        for (int r = 0; r < 8; ++r)
            ldsCarry[8 * h + r] = carry8[r];
    }
    __syncthreads();

    if (tid < 16) {
        float s = 0.0f;
        #pragma unroll
        for (int ww = 0; ww < 8; ++ww)
            s += ldsSum[ww * 16 + tid];
        float logZ = ldsCarry[tid] + fast_ln(s);
        out[b0 + tid] = logZ - gold[b0 + tid];
    }
}

// ---------------------------------------------------------------------------
extern "C" void kernel_launch(void* const* d_in, const int* in_sizes, int n_in,
                              void* d_out, int out_size, void* d_ws, size_t ws_size,
                              hipStream_t stream)
{
    const float* X     = (const float*)d_in[0];
    const int*   y     = (const int*)d_in[1];
    const float* trans = (const float*)d_in[2];
    float*       out   = (float*)d_out;
    float*       gold  = (float*)d_ws;   // NBATCH floats of scratch

    crf_gold_kernel<<<NBATCH, 256, 0, stream>>>(X, y, trans, gold);
    crf_forward_kernel<<<NBATCH / 16, 256, 0, stream>>>(X, trans, gold, out);
}
